// TensorProductScoreModel_14783277432842
// MI455X (gfx1250) — compile-verified, matrix operand order checked by reading, not run on hardware
//
#include <hip/hip_runtime.h>
#include <hip/hip_bf16.h>

typedef __bf16 bf16;
typedef __attribute__((ext_vector_type(16))) __bf16 v16bf;
typedef __attribute__((ext_vector_type(8)))  __bf16 v8bf;
typedef __attribute__((ext_vector_type(8)))  float  v8f;

#define NS 48
#define NV 10
#define IN_DIM 78
#define OUT_DIM 108
#define EDGE_DIM 144
#define KPAD 160
#define WNUM 3564
#define WPAD 3584
#define N_NODES 10000
#define N_EDGES 50000

#define INV_SQRT3 0.57735026918962576f
#define INV_SQRT6 0.40824829046386302f
// C0 = sqrt(1/58), C1 = sqrt(3/68), C2 = sqrt(3/10)
#define C0_ 0.13130643285972254f
#define C1_ 0.21004201260420147f
#define C2_ 0.54772255750516612f

// W121: real 1 (x) 2 -> 1 coupling, unit Frobenius norm, e3nn (y,z,x) / (xy,yz,z2,xz,x2-y2) bases.
// s = 1/sqrt(10), t = 1/sqrt(30)
__constant__ float cW121[3*5*3] = {
  // i=0 (y)
  /*j=0*/ 0.f, 0.f, 0.31622776601f,
  /*j=1*/ 0.f, 0.31622776601f, 0.f,
  /*j=2*/ -0.18257418584f, 0.f, 0.f,
  /*j=3*/ 0.f, 0.f, 0.f,
  /*j=4*/ -0.31622776601f, 0.f, 0.f,
  // i=1 (z)
  /*j=0*/ 0.f, 0.f, 0.f,
  /*j=1*/ 0.31622776601f, 0.f, 0.f,
  /*j=2*/ 0.f, 0.36514837167f, 0.f,
  /*j=3*/ 0.f, 0.f, 0.31622776601f,
  /*j=4*/ 0.f, 0.f, 0.f,
  // i=2 (x)
  /*j=0*/ 0.31622776601f, 0.f, 0.f,
  /*j=1*/ 0.f, 0.f, 0.f,
  /*j=2*/ 0.f, 0.f, -0.18257418584f,
  /*j=3*/ 0.f, 0.31622776601f, 0.f,
  /*j=4*/ 0.f, 0.f, 0.31622776601f,
};

// ---- prep: build bf16 transposed weight matrices  W1T[144][160], W2T[3584][160]
__global__ void prep_weights(const float* __restrict__ fc_w1,
                             const float* __restrict__ fc_w2,
                             bf16* __restrict__ w1t, bf16* __restrict__ w2t) {
  int i = blockIdx.x * 256 + threadIdx.x;
  const int N1 = EDGE_DIM * KPAD;          // 23040
  const int N2 = WPAD * KPAD;              // 573440
  if (i < N1) {
    int n = i / KPAD, k = i % KPAD;
    float v = (k < EDGE_DIM) ? fc_w1[k * EDGE_DIM + n] : 0.f;
    w1t[i] = (bf16)v;
  } else if (i < N1 + N2) {
    int j = i - N1;
    int n = j / KPAD, k = j % KPAD;
    float v = (k < EDGE_DIM && n < WNUM) ? fc_w2[k * WNUM + n] : 0.f;
    w2t[j] = (bf16)v;
  }
}

__global__ void init_zero(float* __restrict__ out, float* __restrict__ cnt) {
  int i = blockIdx.x * 256 + threadIdx.x;
  if (i < N_NODES * OUT_DIM) out[i] = 0.f;
  if (i < N_NODES) cnt[i] = 0.f;
}

// load A operand (M=lm, K-striped by half) from an LDS tile with row stride KPAD
__device__ __forceinline__ v16bf loadA(const bf16* base, int lm, int half, int k0) {
  const bf16* p = base + lm * KPAD + k0 + half * 8;
  v8bf lo = *(const v8bf*)p;
  v8bf hi = *(const v8bf*)(p + 16);
  return __builtin_shufflevector(lo, hi, 0,1,2,3,4,5,6,7,8,9,10,11,12,13,14,15);
}
// load B operand from transposed weights: row n (length KPAD)
__device__ __forceinline__ v16bf loadBT(const bf16* rowp, int half, int k0) {
  const bf16* p = rowp + k0 + half * 8;
  v8bf lo = *(const v8bf*)p;
  v8bf hi = *(const v8bf*)(p + 16);
  return __builtin_shufflevector(lo, hi, 0,1,2,3,4,5,6,7,8,9,10,11,12,13,14,15);
}

__global__ __launch_bounds__(128, 1)
void fused_tp_conv(const float* __restrict__ node_attr,
                   const float* __restrict__ edge_attr,
                   const float* __restrict__ edge_sh,
                   const float* __restrict__ fc_b1,
                   const float* __restrict__ fc_b2,
                   const int*   __restrict__ edge_index,
                   const bf16*  __restrict__ w1t,
                   const bf16*  __restrict__ w2t,
                   float* __restrict__ out_num,
                   float* __restrict__ cnt) {
  __shared__ __align__(16) bf16 sAttr[16][KPAD];
  __shared__ __align__(16) bf16 sH[16][KPAD];
  __shared__ float sX0[16][NS];
  __shared__ float sX1[16][3 * NV];
  __shared__ float sS[16][9];
  __shared__ float sTA[16][NS];
  __shared__ float sTB[16][NV];
  __shared__ float sTE[16][3 * NV];
  __shared__ float sTF[16][3 * NV];
  __shared__ float sAcc0[16][NS];
  __shared__ float sGC[16][NV];
  __shared__ float sGD[16][3 * NV];
  __shared__ float sAE[16][3 * NV];
  __shared__ float sAF[16][3 * NV];
  __shared__ int sSrc[16], sDst[16];

  const int tid  = threadIdx.x;
  const int lane = tid & 31;
  const int wave = tid >> 5;
  const int lm   = lane & 15;
  const int half = lane >> 4;
  const int e0   = blockIdx.x * 16;

  if (tid < 16) {
    sSrc[tid] = edge_index[e0 + tid];
    sDst[tid] = edge_index[N_EDGES + e0 + tid];
  }
  __syncthreads();

  // stage node features of dst, spherical harmonics, edge_attr (bf16), zero pads & accumulators
  for (int i = tid; i < 16 * IN_DIM; i += 128) {
    int m = i / IN_DIM, c = i % IN_DIM;
    float v = node_attr[sDst[m] * IN_DIM + c];
    if (c < NS) sX0[m][c] = v; else sX1[m][c - NS] = v;
  }
  for (int i = tid; i < 16 * 9; i += 128)
    sS[i / 9][i % 9] = edge_sh[(e0 + i / 9) * 9 + (i % 9)];
  for (int i = tid; i < 16 * EDGE_DIM; i += 128) {
    int m = i / EDGE_DIM, c = i % EDGE_DIM;
    sAttr[m][c] = (bf16)edge_attr[(e0 + m) * EDGE_DIM + c];
  }
  for (int i = tid; i < 16 * 16; i += 128) {
    int m = i / 16, c = EDGE_DIM + (i % 16);
    sAttr[m][c] = (bf16)0.f;
    sH[m][c]    = (bf16)0.f;
  }
  for (int i = tid; i < 16 * NS; i += 128) sAcc0[i / NS][i % NS] = 0.f;
  for (int i = tid; i < 16 * NV; i += 128) sGC[i / NV][i % NV] = 0.f;
  for (int i = tid; i < 16 * 3 * NV; i += 128) {
    int m = i / (3 * NV), p = i % (3 * NV);
    sGD[m][p] = 0.f; sAE[m][p] = 0.f; sAF[m][p] = 0.f;
  }
  __syncthreads();

  // t-vectors from Wigner contractions
  for (int i = tid; i < 16 * NS; i += 128) {
    int m = i / NS, u = i % NS;
    sTA[m][u] = sX0[m][u] * sS[m][0];                 // W000 = 1
  }
  for (int i = tid; i < 16 * NV; i += 128) {
    int m = i / NV, u = i % NV;
    const float x0v = sX1[m][u * 3 + 0], x1v = sX1[m][u * 3 + 1], x2v = sX1[m][u * 3 + 2];
    const float s10 = sS[m][1], s11 = sS[m][2], s12 = sS[m][3];
    sTB[m][u] = (x0v * s10 + x1v * s11 + x2v * s12) * INV_SQRT3;      // W110 = delta/sqrt3
    sTF[m][u * 3 + 0] = (x1v * s12 - x2v * s11) * INV_SQRT6;          // W111 = eps/sqrt6
    sTF[m][u * 3 + 1] = (x2v * s10 - x0v * s12) * INV_SQRT6;
    sTF[m][u * 3 + 2] = (x0v * s11 - x1v * s10) * INV_SQRT6;
    float xv[3] = {x0v, x1v, x2v};
    #pragma unroll
    for (int k = 0; k < 3; ++k) {                                     // W121 contraction
      float acc = 0.f;
      #pragma unroll
      for (int ii = 0; ii < 3; ++ii)
        #pragma unroll
        for (int j = 0; j < 5; ++j)
          acc += xv[ii] * sS[m][4 + j] * cW121[(ii * 5 + j) * 3 + k];
      sTE[m][u * 3 + k] = acc;
    }
  }
  __syncthreads();

  // ---- GEMM1: H = relu(Attr @ W1 + b1), bf16 WMMA, K padded to 160
  for (int nt = wave; nt < 9; nt += 4) {
    v8f c = {};
    #pragma unroll
    for (int kk = 0; kk < 5; ++kk) {
      const int k0 = kk * 32;
      v16bf a = loadA(&sAttr[0][0], lm, half, k0);
      v16bf b = loadBT(w1t + (nt * 16 + lm) * KPAD, half, k0);
      c = __builtin_amdgcn_wmma_f32_16x16x32_bf16(false, a, false, b, (short)0, c, false, false);
    }
    const int n = nt * 16 + lm;
    const float bias = fc_b1[n];
    #pragma unroll
    for (int r = 0; r < 8; ++r) {
      const int m = r + 8 * half;
      float h = c[r] + bias;
      sH[m][n] = (bf16)(h > 0.f ? h : 0.f);
    }
  }
  __syncthreads();

  // ---- GEMM2 streamed in 64-wide N chunks, each 16x16 tile applied immediately
  for (int it = 0; it < 56; ++it) {
    const int n = it * 64 + wave * 16 + lm;
    v8f c = {};
    #pragma unroll
    for (int kk = 0; kk < 5; ++kk) {
      const int k0 = kk * 32;
      v16bf a = loadA(&sH[0][0], lm, half, k0);
      v16bf b = loadBT(w2t + n * KPAD, half, k0);
      c = __builtin_amdgcn_wmma_f32_16x16x32_bf16(false, a, false, b, (short)0, c, false, false);
    }
    const float b2 = (n < WNUM) ? fc_b2[n] : 0.f;
    // segment decode (per-lane constant over the 8 rows)
    int seg, u, v;
    if      (n < 2304) { seg = 0; u = n / 48;          v = n - u * 48; }
    else if (n < 2784) { int q = n - 2304; seg = 1; u = q / 10; v = q - u * 10; }
    else if (n < 2884) { int q = n - 2784; seg = 2; u = q / 10; v = q - u * 10; }
    else if (n < 3364) { int q = n - 2884; seg = 3; u = q / 48; v = q - u * 48; }
    else if (n < 3464) { int q = n - 3364; seg = 4; u = q / 10; v = q - u * 10; }
    else if (n < 3564) { int q = n - 3464; seg = 5; u = q / 10; v = q - u * 10; }
    else               { seg = 6; u = 0; v = 0; }
    #pragma unroll
    for (int r = 0; r < 8; ++r) {
      const int m = r + 8 * half;
      const float wv = c[r] + b2;
      switch (seg) {
        case 0: atomicAdd(&sAcc0[m][v], sTA[m][u] * wv); break;            // wA
        case 1: atomicAdd(&sGC[m][v],   sX0[m][u] * wv); break;            // wC
        case 2:                                                            // wD
          atomicAdd(&sGD[m][v * 3 + 0], sX1[m][u * 3 + 0] * wv);
          atomicAdd(&sGD[m][v * 3 + 1], sX1[m][u * 3 + 1] * wv);
          atomicAdd(&sGD[m][v * 3 + 2], sX1[m][u * 3 + 2] * wv); break;
        case 3: atomicAdd(&sAcc0[m][v], sTB[m][u] * wv); break;            // wB
        case 4:                                                            // wF
          atomicAdd(&sAF[m][v * 3 + 0], sTF[m][u * 3 + 0] * wv);
          atomicAdd(&sAF[m][v * 3 + 1], sTF[m][u * 3 + 1] * wv);
          atomicAdd(&sAF[m][v * 3 + 2], sTF[m][u * 3 + 2] * wv); break;
        case 5:                                                            // wE
          atomicAdd(&sAE[m][v * 3 + 0], sTE[m][u * 3 + 0] * wv);
          atomicAdd(&sAE[m][v * 3 + 1], sTE[m][u * 3 + 1] * wv);
          atomicAdd(&sAE[m][v * 3 + 2], sTE[m][u * 3 + 2] * wv); break;
        default: break;                                                    // padding
      }
    }
  }
  __syncthreads();

  // ---- finalize: assemble 108 outputs per edge, scatter-add by src
  for (int i = tid; i < 16 * OUT_DIM; i += 128) {
    const int m = i / OUT_DIM, j = i % OUT_DIM;
    float val;
    if (j < NS) {
      val = C0_ * sAcc0[m][j];
    } else if (j < NS + 3 * NV) {
      const int p = j - NS;
      const int wv = p / 3, k = p - wv * 3;
      val = C1_ * (sGC[m][wv] * sS[m][1 + k] * INV_SQRT3
                 + sS[m][0] * INV_SQRT3 * sGD[m][p]
                 + sAE[m][p]);
    } else {
      val = C2_ * sAF[m][j - NS - 3 * NV];
    }
    atomicAdd(&out_num[sSrc[m] * OUT_DIM + j], val);
  }
  if (tid < 16) atomicAdd(&cnt[sSrc[tid]], 1.0f);
}

__global__ void divide_by_count(float* __restrict__ out, const float* __restrict__ cnt) {
  int i = blockIdx.x * 256 + threadIdx.x;
  if (i < N_NODES * OUT_DIM) {
    float c = cnt[i / OUT_DIM];
    out[i] = out[i] / fmaxf(c, 1.0f);
  }
}

extern "C" void kernel_launch(void* const* d_in, const int* in_sizes, int n_in,
                              void* d_out, int out_size, void* d_ws, size_t ws_size,
                              hipStream_t stream) {
  const float* node_attr = (const float*)d_in[0];
  const float* edge_attr = (const float*)d_in[1];
  const float* edge_sh   = (const float*)d_in[2];
  const float* fc_w1     = (const float*)d_in[3];
  const float* fc_b1     = (const float*)d_in[4];
  const float* fc_w2     = (const float*)d_in[5];
  const float* fc_b2     = (const float*)d_in[6];
  const int*   edge_idx  = (const int*)d_in[7];
  float* out = (float*)d_out;

  char* ws = (char*)d_ws;
  float* cnt = (float*)(ws);                          // 40000 B
  bf16*  w1t = (bf16*)(ws + 65536);                   // 144*160*2 = 46080 B
  bf16*  w2t = (bf16*)(ws + 131072);                  // 3584*160*2 = 1146880 B

  {
    const int total = EDGE_DIM * KPAD + WPAD * KPAD;  // 596480
    prep_weights<<<(total + 255) / 256, 256, 0, stream>>>(fc_w1, fc_w2, w1t, w2t);
  }
  {
    const int total = N_NODES * OUT_DIM;              // also covers cnt
    init_zero<<<(total + 255) / 256, 256, 0, stream>>>(out, cnt);
  }
  fused_tp_conv<<<N_EDGES / 16, 128, 0, stream>>>(node_attr, edge_attr, edge_sh,
                                                  fc_b1, fc_b2, edge_idx,
                                                  w1t, w2t, out, cnt);
  {
    const int total = N_NODES * OUT_DIM;
    divide_by_count<<<(total + 255) / 256, 256, 0, stream>>>(out, cnt);
  }
}
